// MultiHeadLinearCrossAttention_13924283973643
// MI455X (gfx1250) — matrix operand order repaired and checked by reading, not compiled, not run on hardware
//
#include <hip/hip_runtime.h>
#include <hip/hip_bf16.h>

// ---------------------------------------------------------------------------
// MultiHeadLinearCrossAttention for MI455X (gfx1250), wave32 + WMMA.
// B=4, L=S=4096, D=2048, H=16, hd=128.
// Pipeline:
//   1) cvt f32->bf16: query, kv, Wq, Wg, Wkv, Wo
//   2) qf   = fmap(query @ Wq^T)              (bf16 WMMA GEMM, fmap epilogue)
//   3) gate = sigmoid(query @ Wg^T + bg)      (bf16 WMMA GEMM, gate epilogue)
//   4) k,v  = split(kv @ Wkv^T), fmap on k    (bf16 WMMA GEMM, split epilogue)
//   5) ksum = sum_s k                         (VALU reduction)
//   6) S2[e][d] = sum_s v[s,e] k[s,d]         (f32 WMMA 16x16x4, s-major LDS)
//   7) z = 1/(qf . ksum + eps)                (VALU)
//   8) attn_g = (qf @ kv_summary) * z * gate  (bf16 WMMA GEMM)
//   9) out = attn_g @ Wo^T                    (bf16 WMMA GEMM, f32 store)
// GEMM mainloop: double-buffered LDS fed by global_load_async_to_lds_b128
// (ASYNCcnt-tracked), one barrier per K-step, WMMA overlapped with DMA.
// ---------------------------------------------------------------------------

typedef __attribute__((ext_vector_type(16))) __bf16 bf16x16;
typedef __attribute__((ext_vector_type(8)))  float  f32x8;
typedef __attribute__((ext_vector_type(2)))  float  f32x2;

struct U8x32 { uint4 lo, hi; };   // 32 bytes -> bf16x16

static constexpr int CB = 4;
static constexpr int CL = 4096;
static constexpr int CS = 4096;
static constexpr int CD = 2048;
static constexpr int CH = 16;
static constexpr int CHD = 128;

enum { EPI_F32OUT = 0, EPI_FMAP = 1, EPI_GATE = 2, EPI_KVSPLIT = 3, EPI_ATTN = 4 };

// ---- bf16 helpers (RNE) ----------------------------------------------------
__device__ __forceinline__ unsigned short f2bfu(float f) {
  unsigned u = __builtin_bit_cast(unsigned, f);
  u += 0x7FFFu + ((u >> 16) & 1u);
  return (unsigned short)(u >> 16);
}
__device__ __forceinline__ __bf16 f2bf(float f) {
  unsigned short h = f2bfu(f);
  return __builtin_bit_cast(__bf16, h);
}
__device__ __forceinline__ float bfu2f(unsigned short h) {
  unsigned u = ((unsigned)h) << 16;
  return __builtin_bit_cast(float, u);
}
__device__ __forceinline__ float bf2f(__bf16 b) {
  return bfu2f(__builtin_bit_cast(unsigned short, b));
}
__device__ __forceinline__ void cvt8(uint4 u, float* dst) {
  const unsigned x[4] = {u.x, u.y, u.z, u.w};
#pragma unroll
  for (int i = 0; i < 4; ++i) {
    dst[2 * i]     = bfu2f((unsigned short)(x[i] & 0xFFFFu));
    dst[2 * i + 1] = bfu2f((unsigned short)(x[i] >> 16));
  }
}
__device__ __forceinline__ float fmap(float x) {          // elu(x)+1
  return x > 0.f ? x + 1.f : __expf(x);
}

// ---- CDNA5 async global->LDS copy (16B per lane, ASYNCcnt-tracked) ---------
__device__ __forceinline__ void async_ld16(unsigned lds_off, const void* g) {
  asm volatile("global_load_async_to_lds_b128 %0, %1, off"
               :: "v"(lds_off), "v"((unsigned long long)(uintptr_t)g)
               : "memory");
}
__device__ __forceinline__ void wait_async0() {
  asm volatile("s_wait_asynccnt 0" ::: "memory");
}

// ---- elementwise f32 -> bf16 ----------------------------------------------
__global__ void cvt_f32_bf16(const float* __restrict__ x, __bf16* __restrict__ y, long n4) {
  long i  = (long)blockIdx.x * blockDim.x + threadIdx.x;
  long st = (long)gridDim.x * blockDim.x;
  for (; i < n4; i += st) {
    float4 f = ((const float4*)x)[i];
    uint2 o;
    o.x = (unsigned)f2bfu(f.x) | ((unsigned)f2bfu(f.y) << 16);
    o.y = (unsigned)f2bfu(f.z) | ((unsigned)f2bfu(f.w) << 16);
    ((uint2*)y)[i] = o;
  }
}

// ---- generic bf16 WMMA GEMM: C = epi(A @ W^T) ------------------------------
// A:[M,K] bf16 row-major (lda), W:[N,K] bf16 row-major (ldb).
// Tile 128x128x32, 256 threads = 8 waves (2x4), 4x2 16x16x32 frags per wave.
// Double-buffered LDS, async global->LDS staging, one barrier per K-step.
// Per-z offsets: off = (bz/16)*s1 + (bz%16)*s2   (grid.z=1 -> off=0).
template <int EPI>
__global__ __launch_bounds__(256)
void gemm_wmma_bf16(const __bf16* __restrict__ A, long sA1, long sA2, int lda,
                    const __bf16* __restrict__ W, long sB1, long sB2, int ldb,
                    void* __restrict__ Cv, long sC1, long sC2, int ldc,
                    int K,
                    const float* __restrict__ bias,
                    const float* __restrict__ zb, long sZ1, long sZ2,
                    const __bf16* __restrict__ gate,
                    __bf16* __restrict__ C2, int Dcols) {
  __shared__ __bf16 As[2][128 * 40];   // stride 40 (80B) to dodge bank conflicts
  __shared__ __bf16 Bs[2][128 * 40];

  const int tid    = threadIdx.x;
  const int lane   = tid & 31;
  const int wave   = tid >> 5;
  const int wm     = wave >> 2;          // 0..1  (M)
  const int wn     = wave & 3;           // 0..3  (N)
  const int lane15 = lane & 15;
  const int lhalf  = lane >> 4;

  const int bz = blockIdx.z;
  const int zq = bz >> 4, zr = bz & 15;
  const __bf16* Ab = A + zq * sA1 + zr * sA2;
  const __bf16* Wb = W + zq * sB1 + zr * sB2;
  const long coff  = zq * sC1 + zr * sC2;
  const long zoff  = zq * sZ1 + zr * sZ2;

  const int m0 = blockIdx.y * 128;
  const int n0 = blockIdx.x * 128;

  f32x8 acc[4][2];
#pragma unroll
  for (int i = 0; i < 4; ++i)
#pragma unroll
    for (int j = 0; j < 2; ++j)
#pragma unroll
      for (int r = 0; r < 8; ++r) acc[i][j][r] = 0.f;

  const int srow = tid >> 2;             // 0..63
  const int scol = (tid & 3) << 3;       // 0,8,16,24

  auto stage_async = [&](int buf, int k0) {
#pragma unroll
    for (int p = 0; p < 2; ++p) {
      const int r = p * 64 + srow;
      async_ld16((unsigned)(uintptr_t)&As[buf][r * 40 + scol],
                 &Ab[(long)(m0 + r) * lda + k0 + scol]);
      async_ld16((unsigned)(uintptr_t)&Bs[buf][r * 40 + scol],
                 &Wb[(long)(n0 + r) * ldb + k0 + scol]);
    }
  };

  // prologue: fill buffer 0
  stage_async(0, 0);
  wait_async0();
  __syncthreads();

  int buf = 0;
  for (int k0 = 0; k0 < K; k0 += 32) {
    const int nxt = buf ^ 1;
    if (k0 + 32 < K) stage_async(nxt, k0 + 32);   // DMA next tile during math

    bf16x16 af[4], bfv[2];
#pragma unroll
    for (int mt = 0; mt < 4; ++mt) {     // A frag: K = lhalf*8 + {0..7, 16..23}
      const int row = wm * 64 + mt * 16 + lane15;
      U8x32 t;
      t.lo = *(const uint4*)&As[buf][row * 40 + lhalf * 8];
      t.hi = *(const uint4*)&As[buf][row * 40 + lhalf * 8 + 16];
      af[mt] = __builtin_bit_cast(bf16x16, t);
    }
#pragma unroll
    for (int nt = 0; nt < 2; ++nt) {     // B frag: K = lhalf*8 + {0..7, 16..23}
      const int rn = wn * 32 + nt * 16 + lane15;
      U8x32 t;
      t.lo = *(const uint4*)&Bs[buf][rn * 40 + lhalf * 8];
      t.hi = *(const uint4*)&Bs[buf][rn * 40 + lhalf * 8 + 16];
      bfv[nt] = __builtin_bit_cast(bf16x16, t);
    }
#pragma unroll
    for (int mt = 0; mt < 4; ++mt)
#pragma unroll
      for (int nt = 0; nt < 2; ++nt)
        acc[mt][nt] = __builtin_amdgcn_wmma_f32_16x16x32_bf16(
            false, af[mt], false, bfv[nt], (short)0, acc[mt][nt], false, false);

    wait_async0();
    __syncthreads();
    buf = nxt;
  }

  // epilogue
#pragma unroll
  for (int mt = 0; mt < 4; ++mt)
#pragma unroll
    for (int nt = 0; nt < 2; ++nt) {
      f32x8 c = acc[mt][nt];
#pragma unroll
      for (int r = 0; r < 8; ++r) {
        const int row = m0 + wm * 64 + mt * 16 + r + 8 * lhalf;
        const int col = n0 + wn * 32 + nt * 16 + lane15;
        const float v = c[r];
        if constexpr (EPI == EPI_F32OUT) {
          ((float*)Cv)[coff + (long)row * ldc + col] = v;
        } else if constexpr (EPI == EPI_FMAP) {
          ((__bf16*)Cv)[coff + (long)row * ldc + col] = f2bf(fmap(v));
        } else if constexpr (EPI == EPI_GATE) {
          const float g = 1.f / (1.f + __expf(-(v + bias[col])));
          ((__bf16*)Cv)[coff + (long)row * ldc + col] = f2bf(g);
        } else if constexpr (EPI == EPI_KVSPLIT) {
          if (col < Dcols) ((__bf16*)Cv)[(long)row * Dcols + col] = f2bf(fmap(v));
          else             C2[(long)row * Dcols + (col - Dcols)] = f2bf(v);
        } else {                              // EPI_ATTN
          const float z = zb[zoff + row];
          const float g = bf2f(gate[coff + (long)row * ldc + col]);
          ((__bf16*)Cv)[coff + (long)row * ldc + col] = f2bf(v * z * g);
        }
      }
    }
}

// ---- kv_summary via v_wmma_f32_16x16x4_f32 ---------------------------------
// S2[bh][e][d] = sum_s v[b,s,h*128+e] * k[b,s,h*128+d]; one block per (b,h).
__global__ __launch_bounds__(256)
void kv_summary_wmma(const __bf16* __restrict__ kb, const __bf16* __restrict__ vb,
                     __bf16* __restrict__ S2) {
  __shared__ float Ks[32 * 132];
  __shared__ float Vs[32 * 132];

  const int tid    = threadIdx.x;
  const int lane   = tid & 31;
  const int wave   = tid >> 5;
  const int wm     = wave >> 2;
  const int wn     = wave & 3;
  const int lane15 = lane & 15;
  const int lhalf  = lane >> 4;

  const int bh  = blockIdx.x;
  const int b   = bh >> 4;
  const int h   = bh & 15;
  const long base = (long)b * CS * CD + (long)h * CHD;

  f32x8 acc[4][2];
#pragma unroll
  for (int i = 0; i < 4; ++i)
#pragma unroll
    for (int j = 0; j < 2; ++j)
#pragma unroll
      for (int r = 0; r < 8; ++r) acc[i][j][r] = 0.f;

  const int strow = tid >> 3;            // 0..31 (s within tile)
  const int stcol = (tid & 7) << 4;      // 0..112 step 16 (d/e within head)

  for (int s0 = 0; s0 < CS; s0 += 32) {
    const __bf16* kp = kb + base + (long)(s0 + strow) * CD + stcol;
    const __bf16* vp = vb + base + (long)(s0 + strow) * CD + stcol;
    uint4 k0 = ((const uint4*)kp)[0], k1 = ((const uint4*)kp)[1];
    uint4 v0 = ((const uint4*)vp)[0], v1 = ((const uint4*)vp)[1];
    cvt8(k0, &Ks[strow * 132 + stcol]);
    cvt8(k1, &Ks[strow * 132 + stcol + 8]);
    cvt8(v0, &Vs[strow * 132 + stcol]);
    cvt8(v1, &Vs[strow * 132 + stcol + 8]);
    __syncthreads();

#pragma unroll
    for (int ks = 0; ks < 32; ks += 4) {
      f32x2 av[4], bv[2];
#pragma unroll
      for (int mt = 0; mt < 4; ++mt) {           // A = v^T : rows e
        const int e = wm * 64 + mt * 16 + lane15;
        av[mt][0] = Vs[(ks + lhalf * 2 + 0) * 132 + e];
        av[mt][1] = Vs[(ks + lhalf * 2 + 1) * 132 + e];
      }
#pragma unroll
      for (int nt = 0; nt < 2; ++nt) {           // B = k : cols d
        const int d = wn * 32 + nt * 16 + lane15;
        bv[nt][0] = Ks[(ks + lhalf * 2 + 0) * 132 + d];
        bv[nt][1] = Ks[(ks + lhalf * 2 + 1) * 132 + d];
      }
#pragma unroll
      for (int mt = 0; mt < 4; ++mt)
#pragma unroll
        for (int nt = 0; nt < 2; ++nt)
          acc[mt][nt] = __builtin_amdgcn_wmma_f32_16x16x4_f32(
              false, av[mt], false, bv[nt], (short)0, acc[mt][nt], false, false);
    }
    __syncthreads();
  }

  __bf16* out = S2 + (long)bh * CHD * CHD;
#pragma unroll
  for (int mt = 0; mt < 4; ++mt)
#pragma unroll
    for (int nt = 0; nt < 2; ++nt) {
      f32x8 c = acc[mt][nt];
#pragma unroll
      for (int r = 0; r < 8; ++r) {
        const int e = wm * 64 + mt * 16 + r + 8 * lhalf;
        const int d = wn * 32 + nt * 16 + lane15;
        out[e * CHD + d] = f2bf(c[r]);
      }
    }
}

// ---- k_sum[b,d] = sum_s k[b,s,d] -------------------------------------------
__global__ void ksum_kernel(const __bf16* __restrict__ kb, float* __restrict__ ks) {
  const int idx = blockIdx.x * 256 + threadIdx.x;     // B*D
  const int b = idx >> 11;
  const int d = idx & (CD - 1);
  const __bf16* p = kb + (long)b * CS * CD + d;
  float s = 0.f;
  for (int sr = 0; sr < CS; ++sr) s += bf2f(p[(long)sr * CD]);
  ks[idx] = s;
}

// ---- z[b,h,l] = 1/(qf[b,l,h*128:] . ksum[b,h*128:] + eps) ------------------
__global__ void z_kernel(const __bf16* __restrict__ qf, const float* __restrict__ ksum,
                         float* __restrict__ zb) {
  const int idx = blockIdx.x * 256 + threadIdx.x;     // B*H*L
  const int l  = idx & (CL - 1);
  const int bh = idx >> 12;
  const int b = bh >> 4, h = bh & 15;
  const __bf16* qp = qf + (long)b * CL * CD + (long)l * CD + h * CHD;
  const float*  kp = ksum + b * CD + h * CHD;
  float s = 0.f;
  for (int d = 0; d < CHD; d += 8) {
    uint4 u = *(const uint4*)(qp + d);
    float t[8];
    cvt8(u, t);
#pragma unroll
    for (int j = 0; j < 8; ++j) s += t[j] * kp[d + j];
  }
  zb[idx] = 1.f / (s + 1e-6f);
}

// ---------------------------------------------------------------------------
extern "C" void kernel_launch(void* const* d_in, const int* in_sizes, int n_in,
                              void* d_out, int out_size, void* d_ws, size_t ws_size,
                              hipStream_t stream) {
  (void)in_sizes; (void)n_in; (void)out_size; (void)ws_size;

  const float* query = (const float*)d_in[0];
  const float* kvin  = (const float*)d_in[1];
  const float* Wq    = (const float*)d_in[2];
  const float* Wg    = (const float*)d_in[3];
  const float* bg    = (const float*)d_in[4];
  const float* Wkv   = (const float*)d_in[5];
  const float* Wo    = (const float*)d_in[6];
  float* out = (float*)d_out;

  constexpr long NLD = (long)CB * CL * CD;            // 33,554,432 elements
  char* p = (char*)d_ws;
  auto take = [&](long bytes) { char* q = p; p += (bytes + 255) & ~255L; return q; };

  __bf16* qb    = (__bf16*)take(NLD * 2);             // bf16(query) -- reused as attn_g
  __bf16* kvb   = (__bf16*)take(NLD * 2);             // bf16(kv)
  __bf16* kbuf  = (__bf16*)take(NLD * 2);             // fmap(k)
  __bf16* vbuf  = (__bf16*)take(NLD * 2);             // v
  __bf16* qf    = (__bf16*)take(NLD * 2);             // fmap(q)
  __bf16* gateb = (__bf16*)take(NLD * 2);             // sigmoid gate
  __bf16* Wqb   = (__bf16*)take((long)CD * CD * 2);
  __bf16* Wgb   = (__bf16*)take((long)CD * CD * 2);
  __bf16* Wkvb  = (__bf16*)take((long)2 * CD * CD * 2);
  __bf16* Wob   = (__bf16*)take((long)CD * CD * 2);
  __bf16* S2    = (__bf16*)take((long)CB * CH * CHD * CHD * 2);
  float*  ksum  = (float*)take((long)CB * CD * 4);
  float*  zbuf  = (float*)take((long)CB * CH * CL * 4);
  __bf16* attng = qb;                                  // alias: qb dead after gate GEMM

  // 1) conversions
  cvt_f32_bf16<<<4096, 256, 0, stream>>>(query, qb, NLD / 4);
  cvt_f32_bf16<<<4096, 256, 0, stream>>>(kvin, kvb, NLD / 4);
  cvt_f32_bf16<<<1024, 256, 0, stream>>>(Wq, Wqb, (long)CD * CD / 4);
  cvt_f32_bf16<<<1024, 256, 0, stream>>>(Wg, Wgb, (long)CD * CD / 4);
  cvt_f32_bf16<<<2048, 256, 0, stream>>>(Wkv, Wkvb, (long)2 * CD * CD / 4);
  cvt_f32_bf16<<<1024, 256, 0, stream>>>(Wo, Wob, (long)CD * CD / 4);

  const dim3 blk(256, 1, 1);

  // 2) qf = fmap(query @ Wq^T)     M=B*L=16384, N=K=2048
  gemm_wmma_bf16<EPI_FMAP><<<dim3(16, 128, 1), blk, 0, stream>>>(
      qb, 0, 0, CD, Wqb, 0, 0, CD, qf, 0, 0, CD, CD,
      nullptr, nullptr, 0, 0, nullptr, nullptr, 0);

  // 3) gate = sigmoid(query @ Wg^T + bg)
  gemm_wmma_bf16<EPI_GATE><<<dim3(16, 128, 1), blk, 0, stream>>>(
      qb, 0, 0, CD, Wgb, 0, 0, CD, gateb, 0, 0, CD, CD,
      bg, nullptr, 0, 0, nullptr, nullptr, 0);

  // 4) k,v = split(kv @ Wkv^T)     M=B*S=16384, N=4096, K=2048
  gemm_wmma_bf16<EPI_KVSPLIT><<<dim3(32, 128, 1), blk, 0, stream>>>(
      kvb, 0, 0, CD, Wkvb, 0, 0, CD, kbuf, 0, 0, CD, CD,
      nullptr, nullptr, 0, 0, nullptr, vbuf, CD);

  // 5) k_sum
  ksum_kernel<<<dim3(CB * CD / 256, 1, 1), blk, 0, stream>>>(kbuf, ksum);

  // 6) S2[e][d] per (b,h) via f32 WMMA 16x16x4
  kv_summary_wmma<<<dim3(CB * CH, 1, 1), blk, 0, stream>>>(kbuf, vbuf, S2);

  // 7) z
  z_kernel<<<dim3(CB * CH * CL / 256, 1, 1), blk, 0, stream>>>(qf, ksum, zbuf);

  // 8) attn_g = (qf @ kv_summary) * z * gate    per (b,h): M=L, N=K=128
  gemm_wmma_bf16<EPI_ATTN><<<dim3(1, CL / 128, CB * CH), blk, 0, stream>>>(
      qf, (long)CL * CD, CHD, CD,
      S2, (long)CH * CHD * CHD, (long)CHD * CHD, CHD,
      attng, (long)CL * CD, CHD, CD, CHD,
      nullptr, zbuf, (long)CH * CL, (long)CL, gateb, nullptr, 0);

  // 9) out = attn_g @ Wo^T  (f32)
  gemm_wmma_bf16<EPI_F32OUT><<<dim3(16, 128, 1), blk, 0, stream>>>(
      attng, 0, 0, CD, Wob, 0, 0, CD, out, 0, 0, CD, CD,
      nullptr, nullptr, 0, 0, nullptr, nullptr, 0);
}
